// SimpleGCN_45938970198401
// MI455X (gfx1250) — matrix operand order repaired and checked by reading, not compiled, not run on hardware
//
#include <hip/hip_runtime.h>
#include <stdint.h>

#define N_FIN  512
#define F1     256
#define F2     16
#define F3     40
#define TPB    256

typedef __attribute__((ext_vector_type(16))) __bf16 v16bf;
typedef __attribute__((ext_vector_type(8)))  float  v8f;

static __device__ __forceinline__ __bf16 f2bf(float f) {
  union { float f; unsigned u; } v; v.f = f;
  unsigned r = v.u + 0x7FFFu + ((v.u >> 16) & 1u);   // round-to-nearest-even
  unsigned short h = (unsigned short)(r >> 16);
  __bf16 b; __builtin_memcpy(&b, &h, 2);
  return b;
}

// ---------------- degree / dinv ----------------
__global__ void deg_init(float* deg, int n) {
  int i = blockIdx.x * blockDim.x + threadIdx.x;
  if (i < n) deg[i] = 1.0f;                          // self-loop contributes 1
}
__global__ void deg_count(const long long* __restrict__ dst, float* deg, int e) {
  int i = blockIdx.x * blockDim.x + threadIdx.x;
  if (i < e) atomicAdd(&deg[(int)dst[i]], 1.0f);
}
__global__ void deg_rsqrt(float* deg, int n) {
  int i = blockIdx.x * blockDim.x + threadIdx.x;
  if (i < n) deg[i] = rsqrtf(deg[i]);                // deg >= 1 always
}

// ---------------- fragment packing (one-time, f32 -> bf16 WMMA layout) ------
// A layout per (mtile, kbIdx): lane<16 holds K {0..7,16..23} of row tm+l16,
// lane>=16 holds K {8..15,24..31}. Fragment = 16 contiguous bf16 per lane.
__global__ void pack_A(const float* __restrict__ X, __bf16* __restrict__ Xb,
                       int mtiles) {
  long long gid = (long long)blockIdx.x * blockDim.x + threadIdx.x;
  const int lane  = (int)(gid & 31);
  const int kbIdx = (int)((gid >> 5) & 15);
  const int mt    = (int)(gid >> 9);
  if (mt >= mtiles) return;
  const int half = lane >> 4, l16 = lane & 15;
  const float* ap = X + (long long)(mt * 16 + l16) * N_FIN + kbIdx * 32 + half * 8;
  const float4 a0 = *(const float4*)(ap + 0);
  const float4 a1 = *(const float4*)(ap + 4);
  const float4 a2 = *(const float4*)(ap + 16);
  const float4 a3 = *(const float4*)(ap + 20);
  v16bf A;
  A[0]=f2bf(a0.x);  A[1]=f2bf(a0.y);  A[2]=f2bf(a0.z);  A[3]=f2bf(a0.w);
  A[4]=f2bf(a1.x);  A[5]=f2bf(a1.y);  A[6]=f2bf(a1.z);  A[7]=f2bf(a1.w);
  A[8]=f2bf(a2.x);  A[9]=f2bf(a2.y);  A[10]=f2bf(a2.z); A[11]=f2bf(a2.w);
  A[12]=f2bf(a3.x); A[13]=f2bf(a3.y); A[14]=f2bf(a3.z); A[15]=f2bf(a3.w);
  *(v16bf*)(Xb + (((long long)mt * 16 + kbIdx) * 32 + lane) * 16) = A;
}

// B layout per (ntile, kbIdx): lane holds column tn+l16, K {half*8+0..7, 16+half*8+0..7}.
__global__ void pack_B(const float* __restrict__ W, __bf16* __restrict__ Wb) {
  long long gid = (long long)blockIdx.x * blockDim.x + threadIdx.x;
  const int lane  = (int)(gid & 31);
  const int kbIdx = (int)((gid >> 5) & 15);
  const int nt    = (int)(gid >> 9);
  if (nt >= F1 / 16) return;
  const int half = lane >> 4, l16 = lane & 15;
  const float* bp = W + (long long)(kbIdx * 32 + half * 8) * F1 + nt * 16 + l16;
  v16bf B;
#pragma unroll
  for (int e = 0; e < 16; ++e) B[e] = f2bf(bp[(long long)(e + (e >= 8 ? 8 : 0)) * F1]);
  *(v16bf*)(Wb + (((long long)nt * 16 + kbIdx) * 32 + lane) * 16) = B;
}

// ---------------- layer-1 GEMM: H1 = (X @ W1) * dinv[row] -------------------
// One wave computes a 16x64 output strip: 1 A-fragment feeds 4 WMMAs per K-step.
// A and all 4 B fragments are double-buffered so step kb+1's loads overlap the
// 4 WMMAs of step kb (removes the load->wmma s_wait_loadcnt 0x0 serialization).
__global__ __launch_bounds__(TPB)
void gemm1_wmma(const __bf16* __restrict__ Xb, const __bf16* __restrict__ Wb,
                const float* __restrict__ dinv, float* __restrict__ H, int mtiles) {
  const int wave = (int)((blockIdx.x * (long long)blockDim.x + threadIdx.x) >> 5);
  const int lane = threadIdx.x & 31;
  const int nGroups = F1 / 64;                       // 4 strips of 4 tiles
  const int mt = wave / nGroups;
  const int ng = wave % nGroups;
  if (mt >= mtiles) return;                          // whole-wave uniform
  const int half = lane >> 4, l16 = lane & 15;

  const __bf16* ap = Xb + ((long long)mt * 16 * 32 + lane) * 16;
  const __bf16* bp = Wb + (((long long)ng * 4) * 16 * 32 + lane) * 16;
  v8f acc[4] = {v8f{}, v8f{}, v8f{}, v8f{}};

  v16bf Acur = *(const v16bf*)(ap);
  v16bf Bcur[4], Bnxt[4];
  v16bf Anxt = Acur;
#pragma unroll
  for (int t = 0; t < 4; ++t) Bcur[t] = *(const v16bf*)(bp + (long long)t * 8192);

#pragma unroll
  for (int kb = 0; kb < 16; ++kb) {                  // K in steps of 32
    if (kb < 15) {                                   // prefetch next K-step
      Anxt = *(const v16bf*)(ap + (kb + 1) * 512);
#pragma unroll
      for (int t = 0; t < 4; ++t)
        Bnxt[t] = *(const v16bf*)(bp + (long long)t * 8192 + (kb + 1) * 512);
    }
#pragma unroll
    for (int t = 0; t < 4; ++t)
      acc[t] = __builtin_amdgcn_wmma_f32_16x16x32_bf16(
          false, Acur, false, Bcur[t], (short)0, acc[t], false, false);
    Acur = Anxt;
#pragma unroll
    for (int t = 0; t < 4; ++t) Bcur[t] = Bnxt[t];
  }

#pragma unroll
  for (int t = 0; t < 4; ++t) {
#pragma unroll
    for (int v = 0; v < 8; ++v) {
      const long long r = mt * 16 + v + half * 8;    // 32-bit C/D layout
      H[r * F1 + (ng * 4 + t) * 16 + l16] = acc[t][v] * dinv[r];
    }
  }
}

// ---------------- small GEMMs (memory-bound): H = (A @ W) * dinv[row] --------
__global__ void gemm_rowcol(const float* __restrict__ A, const float* __restrict__ W,
                            const float* __restrict__ dinv, float* __restrict__ H,
                            int n, int K, int Fout) {
  long long gid = (long long)blockIdx.x * blockDim.x + threadIdx.x;
  if (gid >= (long long)n * Fout) return;
  const int row = (int)(gid / Fout), col = (int)(gid % Fout);
  const float* a = A + (long long)row * K;
  float s = 0.f;
  for (int k = 0; k < K; ++k) s += a[k] * W[(long long)k * Fout + col];
  H[gid] = s * dinv[row];
}

// ---------------- utility / message-passing ----------------
__global__ void copy_f4(const float4* __restrict__ s, float4* __restrict__ d, long long n4) {
  long long i = (long long)blockIdx.x * blockDim.x + threadIdx.x;
  if (i < n4) d[i] = s[i];
}

// Block stages EPB edges' indices in LDS once; threads fan out over float4 chunks.
__global__ __launch_bounds__(TPB)
void scatter_add4(const long long* __restrict__ src, const long long* __restrict__ dst,
                  const float* __restrict__ Hs, float* __restrict__ Acc,
                  int E, int F) {
  const int chunks = F >> 2;
  const int epb    = TPB / chunks;                   // edges per block
  __shared__ int s_src[TPB], s_dst[TPB];
  const int tid   = threadIdx.x;
  const int ebase = blockIdx.x * epb;
  if (tid < epb) {
    const int e = ebase + tid;
    s_src[tid] = (e < E) ? (int)src[e] : -1;
    s_dst[tid] = (e < E) ? (int)dst[e] : -1;
  }
  __syncthreads();
  const int le = tid / chunks;
  const int c  = tid - le * chunks;
  if (le >= epb || ebase + le >= E) return;
  const int s = s_src[le], d = s_dst[le];
  const float4 v = *(const float4*)(Hs + (long long)s * F + c * 4);
  float* p = Acc + (long long)d * F + c * 4;
  atomicAdd(p + 0, v.x); atomicAdd(p + 1, v.y);
  atomicAdd(p + 2, v.z); atomicAdd(p + 3, v.w);
}

__global__ void finalize(float* __restrict__ Acc, const float* __restrict__ dinv,
                         const float* __restrict__ b, int n, int F, int doRelu) {
  long long gid = (long long)blockIdx.x * blockDim.x + threadIdx.x;
  if (gid >= (long long)n * F) return;
  const int row = (int)(gid / F), col = (int)(gid % F);
  float v = Acc[gid] * dinv[row] + b[col];
  Acc[gid] = doRelu ? fmaxf(v, 0.f) : v;
}

__global__ void log_softmax_rows(const float* __restrict__ H, float* __restrict__ out,
                                 int n, int F) {
  int row = blockIdx.x * blockDim.x + threadIdx.x;
  if (row >= n) return;
  const float* h = H + (long long)row * F;
  float m = -3.4e38f;
  for (int c = 0; c < F; ++c) m = fmaxf(m, h[c]);
  float s = 0.f;
  for (int c = 0; c < F; ++c) s += __expf(h[c] - m);
  const float ls = __logf(s) + m;
  float* o = out + (long long)row * F;
  for (int c = 0; c < F; ++c) o[c] = h[c] - ls;
}

static inline int nblk(long long n, int tpb) { return (int)((n + tpb - 1) / tpb); }

extern "C" void kernel_launch(void* const* d_in, const int* in_sizes, int n_in,
                              void* d_out, int out_size, void* d_ws, size_t ws_size,
                              hipStream_t stream) {
  const float*      X  = (const float*)d_in[0];
  const long long*  EI = (const long long*)d_in[1];
  const float*      W1 = (const float*)d_in[2];
  const float*      b1 = (const float*)d_in[3];
  const float*      W2 = (const float*)d_in[4];
  const float*      b2 = (const float*)d_in[5];
  const float*      W3 = (const float*)d_in[6];
  const float*      b3 = (const float*)d_in[7];
  float*            out = (float*)d_out;

  const int N = in_sizes[0] / N_FIN;                 // 50000
  const int E = in_sizes[1] / 2;                     // 800000
  const int mtiles = N / 16;                         // 3125
  const long long* src = EI;
  const long long* dst = EI + E;

  // ---- workspace layout (~154 MB) ----
  char* ws = (char*)d_ws;
  float*  dinv = (float*)ws;
  size_t  off  = (((size_t)N * 4) + 255) & ~(size_t)255;
  __bf16* Xb   = (__bf16*)(ws + off);                // N*512 bf16 (51.2 MB)
  __bf16* Wb   = Xb + (size_t)N * N_FIN;             // 512*256 bf16
  float*  H1   = (float*)(Wb + (size_t)N_FIN * F1);  // N*F1
  float*  ACC1 = H1 + (size_t)N * F1;                // N*F1 (becomes A1 in place)
  // after layer 1 the Xb region is recycled:
  float*  H2   = (float*)Xb;                         // N*F2
  float*  ACC2 = H2 + (size_t)N * F2;                // N*F2 (becomes A2 in place)
  float*  H3   = ACC2 + (size_t)N * F2;              // N*F3
  float*  ACC3 = H3 + (size_t)N * F3;                // N*F3 (becomes logits in place)

  // ---- dinv = rsqrt(deg) ----
  deg_init <<<nblk(N, TPB), TPB, 0, stream>>>(dinv, N);
  deg_count<<<nblk(E, TPB), TPB, 0, stream>>>(dst, dinv, E);
  deg_rsqrt<<<nblk(N, TPB), TPB, 0, stream>>>(dinv, N);

  // ---- pack operands into bf16 WMMA fragments ----
  pack_A<<<nblk((long long)mtiles * 16 * 32, TPB), TPB, 0, stream>>>(X, Xb, mtiles);
  pack_B<<<nblk((long long)(F1 / 16) * 16 * 32, TPB), TPB, 0, stream>>>(W1, Wb);

  // ---- layer 1: WMMA GEMM, one wave per 16x64 strip ----
  const long long waves1 = (long long)mtiles * (F1 / 64);        // 12500 waves
  gemm1_wmma<<<nblk(waves1 * 32, TPB), TPB, 0, stream>>>(Xb, Wb, dinv, H1, mtiles);
  copy_f4   <<<nblk((long long)N * F1 / 4, TPB), TPB, 0, stream>>>(
      (const float4*)H1, (float4*)ACC1, (long long)N * F1 / 4);  // self-loop init
  scatter_add4<<<nblk(E, TPB / (F1 / 4)) , TPB, 0, stream>>>(src, dst, H1, ACC1, E, F1);
  finalize  <<<nblk((long long)N * F1, TPB), TPB, 0, stream>>>(ACC1, dinv, b1, N, F1, 1);

  // ---- layer 2 ----
  gemm_rowcol<<<nblk((long long)N * F2, TPB), TPB, 0, stream>>>(ACC1, W2, dinv, H2, N, F1, F2);
  copy_f4    <<<nblk((long long)N * F2 / 4, TPB), TPB, 0, stream>>>(
      (const float4*)H2, (float4*)ACC2, (long long)N * F2 / 4);
  scatter_add4<<<nblk(E, TPB / (F2 / 4)), TPB, 0, stream>>>(src, dst, H2, ACC2, E, F2);
  finalize   <<<nblk((long long)N * F2, TPB), TPB, 0, stream>>>(ACC2, dinv, b2, N, F2, 1);

  // ---- layer 3 ----
  gemm_rowcol<<<nblk((long long)N * F3, TPB), TPB, 0, stream>>>(ACC2, W3, dinv, H3, N, F2, F3);
  copy_f4    <<<nblk((long long)N * F3 / 4, TPB), TPB, 0, stream>>>(
      (const float4*)H3, (float4*)ACC3, (long long)N * F3 / 4);
  scatter_add4<<<nblk(E, TPB / (F3 / 4)), TPB, 0, stream>>>(src, dst, H3, ACC3, E, F3);
  finalize   <<<nblk((long long)N * F3, TPB), TPB, 0, stream>>>(ACC3, dinv, b3, N, F3, 0);

  // ---- log-softmax over 40 classes ----
  log_softmax_rows<<<nblk(N, TPB), TPB, 0, stream>>>(ACC3, out, N, F3);
}
